// NonLinearSVM_31963146617428
// MI455X (gfx1250) — matrix operand order repaired and checked by reading, not compiled, not run on hardware
//
#include <hip/hip_runtime.h>

// ---------------------------------------------------------------------------
// RBF-kernel SVM scoring for MI455X (gfx1250, wave32, WMMA).
//   out[b] = sum_s exp(-GAMMA * max(|x_b|^2 + |sv_s|^2 - 2*x_b.sv_s, 0)) * alpha_s
// Cross term via bf16 hi/lo split (3x v_wmma_f32_16x16x32_bf16) -> ~f32 accuracy
// at tensor-core rate. Fast path: one-shot f32->bf16 split into pre-swizzled
// WMMA-layout fragments in workspace, then async (LDS-DMA) double-buffered
// staging in the GEMM kernel. Fragment tiles use a split-half layout so every
// ds_load_b128 is LDS-bank-conflict-free (lane stride 16B).
// ---------------------------------------------------------------------------

#define BATCH   8192
#define NUM_SV  8192
#define DIM     256
#define GAMMA   0.1f
// exp(-g*x) = exp2(x * (-g*log2(e)))
#define KFACT   (-0.14426950408889634f)

typedef __attribute__((ext_vector_type(16))) __bf16 v16bf;
typedef __attribute__((ext_vector_type(8)))  float  v8f;
typedef __attribute__((ext_vector_type(4)))  int    v4i_t;

// CDNA5 async global->LDS copies (ASYNCcnt), if the toolchain exposes them.
#if defined(__has_builtin)
# if __has_builtin(__builtin_amdgcn_global_load_async_to_lds_b128) && \
     __has_builtin(__builtin_amdgcn_s_wait_asynccnt)
#  define HAVE_ASYNC 1
# endif
#endif
#ifndef HAVE_ASYNC
# define HAVE_ASYNC 0
#endif

#if HAVE_ASYNC
// Param 1: global (AS1) pointer to int4 vector; param 2: LDS (AS3) pointer.
#define ASYNC_CP_B128(dst_lds, src_glb)                                  \
  __builtin_amdgcn_global_load_async_to_lds_b128(                        \
      (__attribute__((address_space(1))) v4i_t*)(src_glb),               \
      (__attribute__((address_space(3))) v4i_t*)(dst_lds), 0, 0)
#endif

__device__ __forceinline__ unsigned bf16_rne(float f) {
  unsigned u = __float_as_uint(f);
  return (u + 0x7FFFu + ((u >> 16) & 1u)) >> 16;
}

// Split two f32 into packed-bf16 hi and lo words (element 0 in low half).
__device__ __forceinline__ void split_pair(float f0, float f1, unsigned& hw, unsigned& lw) {
  unsigned h0 = bf16_rne(f0), h1 = bf16_rne(f1);
  float r0 = f0 - __uint_as_float(h0 << 16);
  float r1 = f1 - __uint_as_float(h1 << 16);
  unsigned l0 = bf16_rne(r0), l1 = bf16_rne(r1);
  hw = h0 | (h1 << 16);
  lw = l0 | (l1 << 16);
}

__device__ __forceinline__ v8f wmma_bf16(v16bf a, v16bf b, v8f c) {
  return __builtin_amdgcn_wmma_f32_16x16x32_bf16(false, a, false, b, (short)0, c,
                                                 false, false);
}

__device__ __forceinline__ v16bf frag_from_halves(uint4 p0, uint4 p1) {
  uint4 t2[2] = {p0, p1};
  v16bf r;
  __builtin_memcpy(&r, t2, 32);
  return r;
}

// Fragment-tile indexing (dwords): one 8KB tile = [ks(8)][half(2)][lane(32)][4dw].
//   half 0 = vector elements j=0..7 (dwords 0..3), half 1 = j=8..15.
// A single b128 read at ((ks*2+h)*32+lane)*4 has 16B lane stride -> no bank
// conflicts (lanes 0..15 cover all 64 banks once, lanes 16..31 the next row).
#define FRAG_IDX(ks, h, lane) (((ks) * 2 + (h)) * 128 + (lane) * 4)

// ---------------------------------------------------------------------------
// Kernel 1: per-row squared norms of x (rows 0..8191) and sv (rows 8192..16383).
// One wave per row: 32 lanes x 8 contiguous floats.
// ---------------------------------------------------------------------------
__global__ __launch_bounds__(256) void svm_rbf_norms(const float* __restrict__ x,
                                                     const float* __restrict__ sv,
                                                     float* __restrict__ xsq,
                                                     float* __restrict__ svsq) {
  int gw   = (blockIdx.x * 256 + threadIdx.x) >> 5;
  int lane = threadIdx.x & 31;
  const float* src = (gw < BATCH) ? (x + (size_t)gw * DIM)
                                  : (sv + (size_t)(gw - BATCH) * DIM);
  const float4* p = (const float4*)(src + lane * 8);
  float4 a = p[0], b = p[1];
  float s = a.x * a.x + a.y * a.y + a.z * a.z + a.w * a.w +
            b.x * b.x + b.y * b.y + b.z * b.z + b.w * b.w;
  s += __shfl_xor(s, 1);
  s += __shfl_xor(s, 2);
  s += __shfl_xor(s, 4);
  s += __shfl_xor(s, 8);
  s += __shfl_xor(s, 16);
  if (lane == 0) {
    if (gw < BATCH) xsq[gw] = s;
    else            svsq[gw - BATCH] = s;
  }
}

// ---------------------------------------------------------------------------
// Kernel 1b (fast path): one-shot f32 -> bf16 hi/lo split into PRE-SWIZZLED
// WMMA fragment layout (split-half tiles, see FRAG_IDX).
//   blockIdx.y == 0: X as A-fragments:  [strip(512)][2048 dw tile]
//     A 16x32 bf16 layout: K = ks*32 + (lane>=16 ? 16 : 0) + j
//   blockIdx.y == 1: SV as B-tiles:     [tile(512)][2048 dw tile]
//     B 32x16 bf16 layout: lane l col = l&15, K = ks*32 + (l>=16?16:0) + j
// One thread per (group, ks, lane) produces one 32B hi + 32B lo fragment.
// ---------------------------------------------------------------------------
__global__ __launch_bounds__(512) void svm_rbf_split(const float* __restrict__ x,
                                                     const float* __restrict__ sv,
                                                     unsigned* __restrict__ Xhi,
                                                     unsigned* __restrict__ Xlo,
                                                     unsigned* __restrict__ Bhi,
                                                     unsigned* __restrict__ Blo) {
  int tid  = blockIdx.x * 512 + threadIdx.x;
  int lane = tid & 31;
  int ks   = (tid >> 5) & 7;
  int grp  = tid >> 8;                       // strip / tile index, 0..511
  float s[16];
  unsigned* dhi;
  unsigned* dlo;
  if (blockIdx.y == 0) {
    const float* row = x + (size_t)(grp * 16 + (lane & 15)) * DIM;
    int base = ks * 32 + ((lane & 16) ? 16 : 0);
    float4 f0 = *(const float4*)(row + base);
    float4 f1 = *(const float4*)(row + base + 4);
    float4 f2 = *(const float4*)(row + base + 8);
    float4 f3 = *(const float4*)(row + base + 12);
    s[0]=f0.x; s[1]=f0.y; s[2]=f0.z; s[3]=f0.w; s[4]=f1.x; s[5]=f1.y; s[6]=f1.z; s[7]=f1.w;
    s[8]=f2.x; s[9]=f2.y; s[10]=f2.z; s[11]=f2.w; s[12]=f3.x; s[13]=f3.y; s[14]=f3.z; s[15]=f3.w;
    dhi = Xhi; dlo = Xlo;
  } else {
    const float* row = sv + (size_t)(grp * 16 + (lane & 15)) * DIM;
    int off = ks * 32 + ((lane & 16) ? 16 : 0);
    float4 f0 = *(const float4*)(row + off);
    float4 f1 = *(const float4*)(row + off + 4);
    float4 f2 = *(const float4*)(row + off + 8);
    float4 f3 = *(const float4*)(row + off + 12);
    s[0]=f0.x; s[1]=f0.y; s[2]=f0.z; s[3]=f0.w; s[4]=f1.x; s[5]=f1.y; s[6]=f1.z; s[7]=f1.w;
    s[8]=f2.x; s[9]=f2.y; s[10]=f2.z; s[11]=f2.w; s[12]=f3.x; s[13]=f3.y; s[14]=f3.z; s[15]=f3.w;
    dhi = Bhi; dlo = Blo;
  }
  unsigned hw[8], lw[8];
#pragma unroll
  for (int m = 0; m < 8; ++m) split_pair(s[2 * m], s[2 * m + 1], hw[m], lw[m]);
  size_t o0 = (size_t)grp * 2048 + FRAG_IDX(ks, 0, lane);
  size_t o1 = (size_t)grp * 2048 + FRAG_IDX(ks, 1, lane);
  *(uint4*)(dhi + o0) = make_uint4(hw[0], hw[1], hw[2], hw[3]);
  *(uint4*)(dhi + o1) = make_uint4(hw[4], hw[5], hw[6], hw[7]);
  *(uint4*)(dlo + o0) = make_uint4(lw[0], lw[1], lw[2], lw[3]);
  *(uint4*)(dlo + o1) = make_uint4(lw[4], lw[5], lw[6], lw[7]);
}

// ---------------------------------------------------------------------------
// Kernel 2 (fast path): fused GEMM + exp + weighted row-reduction.
// Block: 512 threads = 16 waves; wave w owns rows [m0 + 16w, m0 + 16w + 16)
// (256 rows/block halves B-side L2 traffic). A fragments (bf16 hi/lo, 128
// VGPRs) stay resident for the whole column chunk. Pre-swizzled B tiles
// (16 KB) are staged with double-buffered async LDS-DMA copies.
// ---------------------------------------------------------------------------
__global__ __launch_bounds__(512) void svm_rbf_main2(const unsigned* __restrict__ Xhi,
                                                     const unsigned* __restrict__ Xlo,
                                                     const unsigned* __restrict__ Bhi,
                                                     const unsigned* __restrict__ Blo,
                                                     const float* __restrict__ dual,
                                                     const float* __restrict__ xsq,
                                                     const float* __restrict__ svsq,
                                                     float* __restrict__ partial,
                                                     int chunkCols) {
  __shared__ unsigned int sB[2][2][2048];   // [buffer][hi/lo][8KB tile]

  const int lane   = threadIdx.x & 31;
  const int wave   = threadIdx.x >> 5;
  const int m0w    = blockIdx.x * 256 + wave * 16;
  const int strip  = blockIdx.x * 16 + wave;
  const int cbase  = blockIdx.y * chunkCols;
  const int tile0  = cbase >> 4;
  const int ntiles = chunkCols >> 4;
  const int hiHalf = (lane >> 4) & 1;
  const int sd     = threadIdx.x * 4;       // this thread's dword slot (16B)

  // Resident A operand: 16 rows x 256 dims, hi+lo bf16 fragments.
  v16bf a_hi[8], a_lo[8];
#pragma unroll
  for (int ks = 0; ks < 8; ++ks) {
    size_t o0 = (size_t)strip * 2048 + FRAG_IDX(ks, 0, lane);
    size_t o1 = (size_t)strip * 2048 + FRAG_IDX(ks, 1, lane);
    a_hi[ks] = frag_from_halves(*(const uint4*)(Xhi + o0), *(const uint4*)(Xhi + o1));
    a_lo[ks] = frag_from_halves(*(const uint4*)(Xlo + o0), *(const uint4*)(Xlo + o1));
  }

  float xs[8];
  {
    const float* xb = xsq + m0w + (hiHalf ? 8 : 0);
    float4 q0 = *(const float4*)xb;
    float4 q1 = *(const float4*)(xb + 4);
    xs[0]=q0.x; xs[1]=q0.y; xs[2]=q0.z; xs[3]=q0.w;
    xs[4]=q1.x; xs[5]=q1.y; xs[6]=q1.z; xs[7]=q1.w;
  }

  float rs[8];
#pragma unroll
  for (int i = 0; i < 8; ++i) rs[i] = 0.0f;

  auto computeTile = [&](int b, int t) {
    v8f acc0 = {};
    v8f acc1 = {};
#pragma unroll
    for (int ks = 0; ks < 8; ++ks) {
      int i0 = FRAG_IDX(ks, 0, lane);
      int i1 = FRAG_IDX(ks, 1, lane);
      v16bf bh = frag_from_halves(*(const uint4*)&sB[b][0][i0],
                                  *(const uint4*)&sB[b][0][i1]);
      v16bf bl = frag_from_halves(*(const uint4*)&sB[b][1][i0],
                                  *(const uint4*)&sB[b][1][i1]);
      acc0 = wmma_bf16(a_hi[ks], bh, acc0);
      acc1 = wmma_bf16(a_lo[ks], bh, acc1);
      acc1 = wmma_bf16(a_hi[ks], bl, acc1);
    }
    int   col = cbase + t * 16 + (lane & 15);
    float svs = svsq[col];
    float dc  = dual[col];
#pragma unroll
    for (int i = 0; i < 8; ++i) {
      float cr = acc0[i] + acc1[i];
      float sq = fmaxf(__builtin_fmaf(-2.0f, cr, xs[i] + svs), 0.0f);
      rs[i] = __builtin_fmaf(__builtin_amdgcn_exp2f(sq * KFACT), dc, rs[i]);
    }
  };

#if HAVE_ASYNC
  // ---- async double-buffered pipeline (ASYNCcnt) --------------------------
  ASYNC_CP_B128(&sB[0][0][sd], Bhi + (size_t)tile0 * 2048 + sd);
  ASYNC_CP_B128(&sB[0][1][sd], Blo + (size_t)tile0 * 2048 + sd);
  for (int t = 0; t < ntiles; ++t) {
    __builtin_amdgcn_s_wait_asynccnt(0);
    __syncthreads();                       // tile t fully in LDS for all waves
    if (t + 1 < ntiles) {
      int nb = (t + 1) & 1;
      size_t go = (size_t)(tile0 + t + 1) * 2048 + sd;
      ASYNC_CP_B128(&sB[nb][0][sd], Bhi + go);
      ASYNC_CP_B128(&sB[nb][1][sd], Blo + go);
    }
    computeTile(t & 1, t);
    __syncthreads();                       // reads done before buffer reuse
  }
#else
  // ---- register-pipelined synchronous fallback ----------------------------
  uint4 rh, rl;
  rh = *(const uint4*)(Bhi + (size_t)tile0 * 2048 + sd);
  rl = *(const uint4*)(Blo + (size_t)tile0 * 2048 + sd);
  for (int t = 0; t < ntiles; ++t) {
    int b = t & 1;
    *(uint4*)&sB[b][0][sd] = rh;
    *(uint4*)&sB[b][1][sd] = rl;
    __syncthreads();
    if (t + 1 < ntiles) {
      size_t go = (size_t)(tile0 + t + 1) * 2048 + sd;
      rh = *(const uint4*)(Bhi + go);
      rl = *(const uint4*)(Blo + go);
    }
    computeTile(b, t);
    __syncthreads();
  }
#endif

  // Reduce across the 16 lanes holding the same rows, then store partials.
#pragma unroll
  for (int i = 0; i < 8; ++i) {
    float v = rs[i];
    v += __shfl_xor(v, 1);
    v += __shfl_xor(v, 2);
    v += __shfl_xor(v, 4);
    v += __shfl_xor(v, 8);
    rs[i] = v;
  }
  if ((lane & 15) == 0) {
    float* dst = partial + (size_t)blockIdx.y * BATCH + m0w + (hiHalf << 3);
    *(float4*)(dst)     = make_float4(rs[0], rs[1], rs[2], rs[3]);
    *(float4*)(dst + 4) = make_float4(rs[4], rs[5], rs[6], rs[7]);
  }
}

// ---------------------------------------------------------------------------
// Kernel 2-fb (small-workspace fallback): split-in-kernel version.
// Block: 256 threads = 8 waves, 128 rows/block. Same split-half LDS layout.
// ---------------------------------------------------------------------------
__global__ __launch_bounds__(256) void svm_rbf_main_fb(const float* __restrict__ x,
                                                       const float* __restrict__ sv,
                                                       const float* __restrict__ dual,
                                                       const float* __restrict__ xsq,
                                                       const float* __restrict__ svsq,
                                                       float* __restrict__ partial,
                                                       int chunkCols) {
  __shared__ unsigned int sBhi[2048];
  __shared__ unsigned int sBlo[2048];

  const int lane   = threadIdx.x & 31;
  const int wave   = threadIdx.x >> 5;
  const int m0w    = blockIdx.x * 128 + wave * 16;
  const int cbase  = blockIdx.y * chunkCols;
  const int hiHalf = (lane >> 4) & 1;

  v16bf a_hi[8], a_lo[8];
  {
    const float* xrow = x + (size_t)(m0w + (lane & 15)) * DIM;
#pragma unroll
    for (int ks = 0; ks < 8; ++ks) {
      int d1 = ks * 32 + (hiHalf ? 16 : 0);
      int d2 = ks * 32 + (hiHalf ? 24 : 8);
      float4 f0 = *(const float4*)(xrow + d1);
      float4 f1 = *(const float4*)(xrow + d1 + 4);
      float4 f2 = *(const float4*)(xrow + d2);
      float4 f3 = *(const float4*)(xrow + d2 + 4);
      unsigned hw[8], lw[8];
      split_pair(f0.x, f0.y, hw[0], lw[0]);
      split_pair(f0.z, f0.w, hw[1], lw[1]);
      split_pair(f1.x, f1.y, hw[2], lw[2]);
      split_pair(f1.z, f1.w, hw[3], lw[3]);
      split_pair(f2.x, f2.y, hw[4], lw[4]);
      split_pair(f2.z, f2.w, hw[5], lw[5]);
      split_pair(f3.x, f3.y, hw[6], lw[6]);
      split_pair(f3.z, f3.w, hw[7], lw[7]);
      __builtin_memcpy(&a_hi[ks], hw, 32);
      __builtin_memcpy(&a_lo[ks], lw, 32);
    }
  }

  float xs[8];
  {
    const float* xb = xsq + m0w + (hiHalf ? 8 : 0);
    float4 q0 = *(const float4*)xb;
    float4 q1 = *(const float4*)(xb + 4);
    xs[0]=q0.x; xs[1]=q0.y; xs[2]=q0.z; xs[3]=q0.w;
    xs[4]=q1.x; xs[5]=q1.y; xs[6]=q1.z; xs[7]=q1.w;
  }

  float rs[8];
#pragma unroll
  for (int i = 0; i < 8; ++i) rs[i] = 0.0f;

  for (int c = 0; c < chunkCols; c += 16) {
    __syncthreads();
    {
      int t    = threadIdx.x;
      int col  = t >> 4;
      int dseg = t & 15;
      int ks   = dseg >> 1;
      int l    = col + ((dseg & 1) << 4);
      const float* svp = sv + (size_t)(cbase + c + col) * DIM + dseg * 16;
      float4 f0 = ((const float4*)svp)[0];
      float4 f1 = ((const float4*)svp)[1];
      float4 f2 = ((const float4*)svp)[2];
      float4 f3 = ((const float4*)svp)[3];
      unsigned hw[8], lw[8];
      split_pair(f0.x, f0.y, hw[0], lw[0]);
      split_pair(f0.z, f0.w, hw[1], lw[1]);
      split_pair(f1.x, f1.y, hw[2], lw[2]);
      split_pair(f1.z, f1.w, hw[3], lw[3]);
      split_pair(f2.x, f2.y, hw[4], lw[4]);
      split_pair(f2.z, f2.w, hw[5], lw[5]);
      split_pair(f3.x, f3.y, hw[6], lw[6]);
      split_pair(f3.z, f3.w, hw[7], lw[7]);
      int i0 = FRAG_IDX(ks, 0, l);
      int i1 = FRAG_IDX(ks, 1, l);
      *(uint4*)&sBhi[i0] = make_uint4(hw[0], hw[1], hw[2], hw[3]);
      *(uint4*)&sBhi[i1] = make_uint4(hw[4], hw[5], hw[6], hw[7]);
      *(uint4*)&sBlo[i0] = make_uint4(lw[0], lw[1], lw[2], lw[3]);
      *(uint4*)&sBlo[i1] = make_uint4(lw[4], lw[5], lw[6], lw[7]);
    }
    __syncthreads();

    v8f acc0 = {};
    v8f acc1 = {};
#pragma unroll
    for (int ks = 0; ks < 8; ++ks) {
      int i0 = FRAG_IDX(ks, 0, lane);
      int i1 = FRAG_IDX(ks, 1, lane);
      v16bf bh = frag_from_halves(*(const uint4*)&sBhi[i0], *(const uint4*)&sBhi[i1]);
      v16bf bl = frag_from_halves(*(const uint4*)&sBlo[i0], *(const uint4*)&sBlo[i1]);
      acc0 = wmma_bf16(a_hi[ks], bh, acc0);
      acc1 = wmma_bf16(a_lo[ks], bh, acc1);
      acc1 = wmma_bf16(a_hi[ks], bl, acc1);
    }

    int   col = cbase + c + (lane & 15);
    float svs = svsq[col];
    float dc  = dual[col];
#pragma unroll
    for (int i = 0; i < 8; ++i) {
      float cr = acc0[i] + acc1[i];
      float sq = fmaxf(__builtin_fmaf(-2.0f, cr, xs[i] + svs), 0.0f);
      rs[i] = __builtin_fmaf(__builtin_amdgcn_exp2f(sq * KFACT), dc, rs[i]);
    }
  }

#pragma unroll
  for (int i = 0; i < 8; ++i) {
    float v = rs[i];
    v += __shfl_xor(v, 1);
    v += __shfl_xor(v, 2);
    v += __shfl_xor(v, 4);
    v += __shfl_xor(v, 8);
    rs[i] = v;
  }
  if ((lane & 15) == 0) {
    float* dst = partial + (size_t)blockIdx.y * BATCH + m0w + (hiHalf << 3);
    *(float4*)(dst)     = make_float4(rs[0], rs[1], rs[2], rs[3]);
    *(float4*)(dst + 4) = make_float4(rs[4], rs[5], rs[6], rs[7]);
  }
}

// ---------------------------------------------------------------------------
// Kernel 3: deterministic reduction of per-chunk partials.
// ---------------------------------------------------------------------------
__global__ __launch_bounds__(256) void svm_rbf_reduce(const float* __restrict__ partial,
                                                      float* __restrict__ out,
                                                      int nchunk) {
  int r = blockIdx.x * blockDim.x + threadIdx.x;
  float s = 0.0f;
  for (int c = 0; c < nchunk; ++c) s += partial[(size_t)c * BATCH + r];
  out[r] = s;
}

// ---------------------------------------------------------------------------
extern "C" void kernel_launch(void* const* d_in, const int* in_sizes, int n_in,
                              void* d_out, int out_size, void* d_ws, size_t ws_size,
                              hipStream_t stream) {
  const float* x    = (const float*)d_in[0];
  const float* sv   = (const float*)d_in[1];
  const float* dual = (const float*)d_in[2];
  float* out = (float*)d_out;
  float* ws  = (float*)d_ws;

  float* xsq  = ws;                 // [8192]
  float* svsq = ws + BATCH;         // [8192]
  float* partial = ws + 2 * BATCH;  // [nchunk][8192]

  const size_t FRAG_DW = 1048576;   // dwords per bf16 fragment array (4 MB)

  // Fast path: needs norms + partials + 4 x 4MB bf16 fragment arrays.
  int  nchunk = 16;
  bool fast   = false;
  for (int nc = 16; nc >= 1; nc >>= 1) {
    size_t need = ((size_t)2 * BATCH + (size_t)nc * BATCH + 4 * FRAG_DW) * sizeof(float);
    if (ws_size >= need) { nchunk = nc; fast = true; break; }
  }

  svm_rbf_norms<<<(BATCH + NUM_SV) / 8, 256, 0, stream>>>(x, sv, xsq, svsq);

  if (fast) {
    unsigned* base = (unsigned*)(ws + 2 * BATCH + (size_t)nchunk * BATCH);
    unsigned* Xhi = base;
    unsigned* Xlo = base + FRAG_DW;
    unsigned* Bhi = base + 2 * FRAG_DW;
    unsigned* Blo = base + 3 * FRAG_DW;
    int chunkCols = NUM_SV / nchunk;

    dim3 sgrid(256, 2);
    svm_rbf_split<<<sgrid, 512, 0, stream>>>(x, sv, Xhi, Xlo, Bhi, Blo);

    dim3 grid(BATCH / 256, nchunk);
    svm_rbf_main2<<<grid, 512, 0, stream>>>(Xhi, Xlo, Bhi, Blo, dual, xsq, svsq,
                                            partial, chunkCols);
    svm_rbf_reduce<<<BATCH / 256, 256, 0, stream>>>(partial, out, nchunk);
  } else {
    int nc = 16;
    while (nc > 1 &&
           ws_size < (size_t)(2 * BATCH + (size_t)nc * BATCH) * sizeof(float))
      nc >>= 1;
    int chunkCols = NUM_SV / nc;
    dim3 grid(BATCH / 128, nc);
    svm_rbf_main_fb<<<grid, 256, 0, stream>>>(x, sv, dual, xsq, svsq, partial,
                                              chunkCols);
    svm_rbf_reduce<<<BATCH / 256, 256, 0, stream>>>(partial, out, nc);
  }
}
